// Attention_34531537059864
// MI455X (gfx1250) — compile-verified, matrix operand order checked by reading, not run on hardware
//
#include <hip/hip_runtime.h>
#include <hip/hip_bf16.h>

// ---------------- problem constants (match reference) ----------------
#define BB   2
#define SS   2048
#define NXC  1024
#define NHC  16
#define HDC  64
#define NCOL (3 * NXC)

// ---------------- feature probes (compile-safe) ----------------
#if defined(__has_builtin)
#  if __has_builtin(__builtin_amdgcn_global_load_async_to_lds_b128) && \
      __has_builtin(__builtin_amdgcn_s_wait_asynccnt)
#    define HAVE_ASYNC_LDS 1
#  endif
#endif
#ifndef HAVE_ASYNC_LDS
#  define HAVE_ASYNC_LDS 0
#endif

#if HAVE_ASYNC_LDS
#  define WAIT_ASYNC(n) __builtin_amdgcn_s_wait_asynccnt(n)
#else
#  define WAIT_ASYNC(n)
#endif

// ---------------- CDNA5 WMMA types ----------------
typedef __attribute__((ext_vector_type(16))) __bf16        v16bf;
typedef __attribute__((ext_vector_type(2)))  __bf16        v2bf;
typedef __attribute__((ext_vector_type(8)))  float         v8f;
typedef __attribute__((ext_vector_type(2)))  float         v2f;
typedef __attribute__((ext_vector_type(8)))  unsigned int  v8u;
typedef __attribute__((ext_vector_type(4)))  int           v4i;

// Compiler-native f32 -> bf16 (fptrunc, RNE); backend picks the best
// instruction sequence for gfx1250.
__device__ __forceinline__ unsigned short f2bf(float x) {
    __bf16 h = (__bf16)x;
    return __builtin_bit_cast(unsigned short, h);
}
__device__ __forceinline__ unsigned int pk2(float a, float b) {
    v2f t;
    t[0] = a; t[1] = b;
    v2bf r = __builtin_convertvector(t, v2bf);
    return __builtin_bit_cast(unsigned int, r);
}

// 16B global -> LDS copy; async (no VGPR round trip, ASYNCcnt) when available.
__device__ __forceinline__ void async_cp16(const float* g, float* l) {
#if HAVE_ASYNC_LDS
    // builtin prototype: (int4 AS1*, int4 AS3*, imm offset, imm cpol)
    __builtin_amdgcn_global_load_async_to_lds_b128(
        (__attribute__((address_space(1))) v4i*)(float*)g,
        (__attribute__((address_space(3))) v4i*)l, 0, 0);
#else
    *reinterpret_cast<uint4*>(l) = *reinterpret_cast<const uint4*>(g);
#endif
}

__device__ __forceinline__ v8f wmma_bf16(v16bf a, v16bf b, v8f c) {
    return __builtin_amdgcn_wmma_f32_16x16x32_bf16(false, a, false, b,
                                                   (short)0, c, false, false);
}

// bf16-LDS fragment: element e <-> k = kbase + (e<8?0:16) + 8*half + e%8
// -> two contiguous 16B runs -> two ds_load_b128.
__device__ __forceinline__ v16bf ld_frag(const unsigned short* lds, int row,
                                         int stride, int kbase, int half) {
    const uint4* p0 = reinterpret_cast<const uint4*>(lds + row * stride + kbase + 8 * half);
    const uint4* p1 = reinterpret_cast<const uint4*>(lds + row * stride + kbase + 16 + 8 * half);
    uint4 a = *p0, b = *p1;
    v8u u;
    u[0] = a.x; u[1] = a.y; u[2] = a.z; u[3] = a.w;
    u[4] = b.x; u[5] = b.y; u[6] = b.z; u[7] = b.w;
    return __builtin_bit_cast(v16bf, u);
}

// f32-LDS fragment (async-staged tiles): 4x ds_load_b128 + convert.
__device__ __forceinline__ v16bf ld_frag_f32(const float* lds, int row,
                                             int stride, int kbase, int half) {
    const float4* p0 = reinterpret_cast<const float4*>(lds + row * stride + kbase + 8 * half);
    const float4* p1 = reinterpret_cast<const float4*>(lds + row * stride + kbase + 16 + 8 * half);
    float4 a = p0[0], b = p0[1], c = p1[0], d = p1[1];
    v8u u;
    u[0] = pk2(a.x, a.y); u[1] = pk2(a.z, a.w);
    u[2] = pk2(b.x, b.y); u[3] = pk2(b.z, b.w);
    u[4] = pk2(c.x, c.y); u[5] = pk2(c.z, c.w);
    u[6] = pk2(d.x, d.y); u[7] = pk2(d.z, d.w);
    return __builtin_bit_cast(v16bf, u);
}

// 16-lane (half-wave) reductions: wave32 C-layout keeps one matrix row per
// VGPR within each 16-lane half.
__device__ __forceinline__ float rmax16(float v) {
#pragma unroll
    for (int m = 1; m < 16; m <<= 1) v = fmaxf(v, __shfl_xor(v, m, 16));
    return v;
}
__device__ __forceinline__ float rsum16(float v) {
#pragma unroll
    for (int m = 1; m < 16; m <<= 1) v += __shfl_xor(v, m, 16);
    return v;
}

// =====================================================================
// Kernel 1: QKV GEMM.  C[4096,3072] = X[4096,1024]*W + b.
// X tile async-staged f32 (double buffered); W tile sync-staged bf16
// transposed.  Scatter epilogue: Q -> q_ws, K/V -> `present` output.
// =====================================================================
#define TM     128
#define TN     64
#define TK     32
#define AF32ST 36   // f32 staging stride (32 + 4 pad)
#define BSTR   40   // bf16 stride (32 + 8 pad)

__global__ __launch_bounds__(256) void qkv_gemm_kernel(
    const float* __restrict__ X, const float* __restrict__ W,
    const float* __restrict__ bias, float* __restrict__ q_ws,
    float* __restrict__ kv_out) {
    __shared__ float          Ast[2][TM * AF32ST];  // X tile f32, ping-pong
    __shared__ unsigned short Bsm[TN * BSTR];       // W^T tile bf16

    const int tid  = threadIdx.x;
    const int wave = tid >> 5;
    const int lane = tid & 31;
    const int l16  = lane & 15;
    const int half = lane >> 4;
    const int n0   = blockIdx.x * TN;
    const int m0   = blockIdx.y * TM;

    auto stage_a = [&](int kb, int p) {   // 128x32 f32, 16B chunks, async
#pragma unroll
        for (int q = tid; q < 1024; q += 256) {
            int row = q >> 3, c4 = (q & 7) * 4;
            async_cp16(X + (size_t)(m0 + row) * NXC + kb + c4,
                       &Ast[p][row * AF32ST + c4]);
        }
    };

    v8f acc[4] = {};
    stage_a(0, 0);
    int p = 0;

    for (int kb = 0; kb < NXC; kb += TK) {
        __syncthreads();                       // prev compute done: Bsm/Ast free
        {   // W tile 32x64 -> bf16 transposed into Bsm[n][k]
            int col = (tid & 15) * 4;
            int rb  = tid >> 4;
#pragma unroll
            for (int pp = 0; pp < 2; pp++) {
                int k = rb + 16 * pp;
                float4 v = *reinterpret_cast<const float4*>(
                    W + (size_t)(kb + k) * NCOL + n0 + col);
                Bsm[(col + 0) * BSTR + k] = f2bf(v.x);
                Bsm[(col + 1) * BSTR + k] = f2bf(v.y);
                Bsm[(col + 2) * BSTR + k] = f2bf(v.z);
                Bsm[(col + 3) * BSTR + k] = f2bf(v.w);
            }
        }
        if (kb + TK < NXC) {                   // pipeline next X tile
            stage_a(kb + TK, p ^ 1);
            __builtin_prefetch(W + (size_t)(kb + TK) * NCOL + n0 + (tid & 15) * 4, 0, 1);
            WAIT_ASYNC(4);                     // tile kb arrived, next in flight
        } else {
            WAIT_ASYNC(0);
        }
        __syncthreads();

        v16bf a = ld_frag_f32(&Ast[p][0], wave * 16 + l16, AF32ST, 0, half);
#pragma unroll
        for (int t = 0; t < 4; t++) {
            v16bf b = ld_frag(Bsm, t * 16 + l16, BSTR, 0, half);
            acc[t]  = wmma_bf16(a, b, acc[t]);
        }
        p ^= 1;
    }

    const int row0 = m0 + wave * 16 + 8 * half;
#pragma unroll
    for (int t = 0; t < 4; t++) {
        int   col = n0 + t * 16 + l16;
        float bv  = bias[col];
        int   sec = col >> 10;
        int   c   = col & 1023;
        int   h   = c >> 6, d = c & 63;
#pragma unroll
        for (int j = 0; j < 8; j++) {
            int    r   = row0 + j;
            int    b   = r >> 11;
            int    s   = r & 2047;
            size_t idx = (((size_t)b * NHC + h) * SS + s) * HDC + d;
            float  val = acc[t][j] + bv;
            if (sec == 0)
                q_ws[idx] = val;                                   // Q
            else if (sec == 1)
                kv_out[idx] = val;                                 // present K
            else
                kv_out[(size_t)BB * NHC * SS * HDC + idx] = val;   // present V
        }
    }
}

// =====================================================================
// Kernel 2: flash attention.  K/V tiles async-staged f32 (double
// buffered); scores fragments convert from f32 LDS on the fly; V gets a
// small LDS->LDS transpose/convert pass.  Scores never touch HBM.
// =====================================================================
#define KT     32   // keys per tile
#define KF32ST 68   // f32 staging stride for [key][64 d] (+4 pad)
#define VSTR   40   // V^T bf16 [d][key] stride
#define PSTR   40   // per-wave P scratch [16 q][32 key] stride

__global__ __launch_bounds__(256) void flash_attn_kernel(
    const float* __restrict__ q_ws, const float* __restrict__ kv,
    float* __restrict__ attn_ws) {
    __shared__ float          Kst[2][KT * KF32ST];
    __shared__ float          Vst[2][KT * KF32ST];
    __shared__ unsigned short Vsm[HDC * VSTR];
    __shared__ unsigned short Psm[8 * 16 * PSTR];

    const int tid  = threadIdx.x;
    const int wave = tid >> 5;
    const int lane = tid & 31;
    const int l16  = lane & 15;
    const int half = lane >> 4;

    const int qc = blockIdx.x % (SS / 128);
    const int bh = blockIdx.x / (SS / 128);       // b*NH + h
    const int qr = qc * 128 + wave * 16;

    const float* qbase = q_ws + (size_t)bh * SS * HDC;
    const float* kg    = kv + (size_t)bh * SS * HDC;
    const float* vg    = kv + (size_t)BB * NHC * SS * HDC + (size_t)bh * SS * HDC;

    auto stage_kv = [&](int kb, int p) {   // 2x (32x64 f32), 16B chunks
#pragma unroll
        for (int q = tid; q < 512; q += 256) {
            int row = q >> 4, c4 = (q & 15) * 4;
            async_cp16(kg + (size_t)(kb + row) * HDC + c4,
                       &Kst[p][row * KF32ST + c4]);
            async_cp16(vg + (size_t)(kb + row) * HDC + c4,
                       &Vst[p][row * KF32ST + c4]);
        }
    };

    // preload this wave's Q tile as two bf16 A-fragments (d 0..31, 32..63)
    const float* qrow = qbase + (size_t)(qr + l16) * HDC;
    v16bf qf[2];
#pragma unroll
    for (int kk = 0; kk < 2; kk++) {
        const float* a = qrow + 32 * kk + 8 * half;
        const float* b = a + 16;
        v8u u;
        u[0] = pk2(a[0], a[1]); u[1] = pk2(a[2], a[3]);
        u[2] = pk2(a[4], a[5]); u[3] = pk2(a[6], a[7]);
        u[4] = pk2(b[0], b[1]); u[5] = pk2(b[2], b[3]);
        u[6] = pk2(b[4], b[5]); u[7] = pk2(b[6], b[7]);
        qf[kk] = __builtin_bit_cast(v16bf, u);
    }

    float M[8], L[8];
#pragma unroll
    for (int j = 0; j < 8; j++) { M[j] = -3.0e38f; L[j] = 0.0f; }
    v8f o[4] = {};
    unsigned short* Pw = Psm + wave * 16 * PSTR;

    stage_kv(0, 0);
    int p = 0;

    for (int kb = 0; kb < SS; kb += KT) {
        __syncthreads();                       // prev compute done: buf p^1 free
        if (kb + KT < SS) {
            stage_kv(kb + KT, p ^ 1);
            WAIT_ASYNC(4);                     // tile kb arrived, next in flight
        } else {
            WAIT_ASYNC(0);
        }
        __syncthreads();

        {   // V: f32 [key][d] -> bf16 transposed [d][key]
            int d4 = (tid & 15) * 4;
            int kq = tid >> 4;
#pragma unroll
            for (int pp = 0; pp < 2; pp++) {
                int key = kq + 16 * pp;
                float4 v = *reinterpret_cast<const float4*>(
                    &Vst[p][key * KF32ST + d4]);
                Vsm[(d4 + 0) * VSTR + key] = f2bf(v.x);
                Vsm[(d4 + 1) * VSTR + key] = f2bf(v.y);
                Vsm[(d4 + 2) * VSTR + key] = f2bf(v.z);
                Vsm[(d4 + 3) * VSTR + key] = f2bf(v.w);
            }
        }
        __syncthreads();

        // ---- scores S = Q K^T / sqrt(64): 16 x 32 per wave ----
        v8f s0 = {}, s1 = {};
#pragma unroll
        for (int kk = 0; kk < 2; kk++) {
            v16bf b0 = ld_frag_f32(&Kst[p][0], l16,      KF32ST, 32 * kk, half);
            v16bf b1 = ld_frag_f32(&Kst[p][0], 16 + l16, KF32ST, 32 * kk, half);
            s0 = wmma_bf16(qf[kk], b0, s0);
            s1 = wmma_bf16(qf[kk], b1, s1);
        }

        // ---- online softmax (row == one VGPR per 16-lane half) ----
#pragma unroll
        for (int j = 0; j < 8; j++) {
            float a0 = s0[j] * 0.125f;
            float a1 = s1[j] * 0.125f;
            float nm = fmaxf(M[j], rmax16(fmaxf(a0, a1)));
            float fc = exp2f((M[j] - nm) * 1.44269504f);
            float p0 = exp2f((a0 - nm) * 1.44269504f);
            float p1 = exp2f((a1 - nm) * 1.44269504f);
            L[j] = L[j] * fc + rsum16(p0 + p1);
            M[j] = nm;
#pragma unroll
            for (int t = 0; t < 4; t++) o[t][j] *= fc;
            int row = j + 8 * half;                 // C-layout -> A-layout
            Pw[row * PSTR + l16]      = f2bf(p0);
            Pw[row * PSTR + 16 + l16] = f2bf(p1);
        }

        // ---- O += P V (contraction over 32 keys = one WMMA k-step) ----
        v16bf pf = ld_frag(Pw, l16, PSTR, 0, half);
#pragma unroll
        for (int t = 0; t < 4; t++) {
            v16bf bv = ld_frag(Vsm, t * 16 + l16, VSTR, 0, half);
            o[t] = wmma_bf16(pf, bv, o[t]);
        }
        p ^= 1;
    }

    // ---- normalize, write [B,S,NX] (row = b*S+s, col = h*64+d) ----
    const int b = bh / NHC, h = bh % NHC;
#pragma unroll
    for (int j = 0; j < 8; j++) {
        int    row  = qr + j + 8 * half;
        float  inv  = 1.0f / L[j];
        size_t base = ((size_t)b * SS + row) * NXC + h * HDC;
#pragma unroll
        for (int t = 0; t < 4; t++)
            attn_ws[base + t * 16 + l16] = o[t][j] * inv;
    }
}

// =====================================================================
// Kernel 3: output projection.  out[4096,1024] = A*Wp + bp.
// =====================================================================
__global__ __launch_bounds__(256) void proj_gemm_kernel(
    const float* __restrict__ A, const float* __restrict__ W,
    const float* __restrict__ bias, float* __restrict__ C) {
    __shared__ float          Ast[2][TM * AF32ST];
    __shared__ unsigned short Bsm[TN * BSTR];

    const int tid  = threadIdx.x;
    const int wave = tid >> 5;
    const int lane = tid & 31;
    const int l16  = lane & 15;
    const int half = lane >> 4;
    const int n0   = blockIdx.x * TN;
    const int m0   = blockIdx.y * TM;

    auto stage_a = [&](int kb, int p) {
#pragma unroll
        for (int q = tid; q < 1024; q += 256) {
            int row = q >> 3, c4 = (q & 7) * 4;
            async_cp16(A + (size_t)(m0 + row) * NXC + kb + c4,
                       &Ast[p][row * AF32ST + c4]);
        }
    };

    v8f acc[4] = {};
    stage_a(0, 0);
    int p = 0;

    for (int kb = 0; kb < NXC; kb += TK) {
        __syncthreads();
        {
            int col = (tid & 15) * 4;
            int rb  = tid >> 4;
#pragma unroll
            for (int pp = 0; pp < 2; pp++) {
                int k = rb + 16 * pp;
                float4 v = *reinterpret_cast<const float4*>(
                    W + (size_t)(kb + k) * NXC + n0 + col);
                Bsm[(col + 0) * BSTR + k] = f2bf(v.x);
                Bsm[(col + 1) * BSTR + k] = f2bf(v.y);
                Bsm[(col + 2) * BSTR + k] = f2bf(v.z);
                Bsm[(col + 3) * BSTR + k] = f2bf(v.w);
            }
        }
        if (kb + TK < NXC) {
            stage_a(kb + TK, p ^ 1);
            __builtin_prefetch(W + (size_t)(kb + TK) * NXC + n0 + (tid & 15) * 4, 0, 1);
            WAIT_ASYNC(4);
        } else {
            WAIT_ASYNC(0);
        }
        __syncthreads();

        v16bf a = ld_frag_f32(&Ast[p][0], wave * 16 + l16, AF32ST, 0, half);
#pragma unroll
        for (int t = 0; t < 4; t++) {
            v16bf b = ld_frag(Bsm, t * 16 + l16, BSTR, 0, half);
            acc[t]  = wmma_bf16(a, b, acc[t]);
        }
        p ^= 1;
    }

    const int row0 = m0 + wave * 16 + 8 * half;
#pragma unroll
    for (int t = 0; t < 4; t++) {
        int   col = n0 + t * 16 + l16;
        float bv  = bias[col];
#pragma unroll
        for (int j = 0; j < 8; j++)
            C[(size_t)(row0 + j) * NXC + col] = acc[t][j] + bv;
    }
}

// =====================================================================
extern "C" void kernel_launch(void* const* d_in, const int* in_sizes, int n_in,
                              void* d_out, int out_size, void* d_ws,
                              size_t ws_size, hipStream_t stream) {
    const float* x      = (const float*)d_in[0];
    const float* w_attn = (const float*)d_in[1];
    const float* b_attn = (const float*)d_in[2];
    const float* w_proj = (const float*)d_in[3];
    const float* b_proj = (const float*)d_in[4];

    float* out     = (float*)d_out;                         // [B,S,NX]
    float* present = out + (size_t)BB * SS * NXC;           // [2,B,H,S,D]
    float* q_ws    = (float*)d_ws;                          // [B,H,S,D] f32
    float* attn_ws = q_ws + (size_t)BB * NHC * SS * HDC;    // [B*S, NX] f32

    qkv_gemm_kernel<<<dim3(NCOL / TN, (BB * SS) / TM), 256, 0, stream>>>(
        x, w_attn, b_attn, q_ws, present);

    flash_attn_kernel<<<dim3(BB * NHC * (SS / 128)), 256, 0, stream>>>(
        q_ws, present, attn_ws);

    proj_gemm_kernel<<<dim3(NXC / TN, (BB * SS) / TM), 256, 0, stream>>>(
        attn_ws, w_proj, b_proj, out);
}
